// MultiheadSelfAttention_40853728920052
// MI455X (gfx1250) — matrix operand order repaired and checked by reading, not compiled, not run on hardware
//
#include <hip/hip_runtime.h>

typedef __attribute__((ext_vector_type(16))) _Float16 v16h;
typedef __attribute__((ext_vector_type(8)))  _Float16 v8h;
typedef __attribute__((ext_vector_type(8)))  float    v8f;

#define B_    2
#define T_    2048
#define M_    2048
#define H_    16
#define D_    128
#define HALF_ 64
#define BT_   (B_ * T_)
#define HD_   (H_ * D_)

// ---------------------------------------------------------------------------
// WMMA fragment helpers (CDNA5 v_wmma_f32_16x16x32_f16, wave32)
// A as loaded here (16x32 f16): lanes 0-15 hold row M=lane, K0..15 contiguous;
// lanes 16-31 hold K16..31.
// B as loaded here, same k order as A (32x16 f16): lanes 0-15 hold col N=lane, K0..15 contiguous;
// lanes 16-31 hold K16..31.
// ---------------------------------------------------------------------------
__device__ __forceinline__ v16h load_a16(const _Float16* __restrict__ row, int k0, int lh) {
  return *(const v16h*)(row + k0 + lh * 16);
}
__device__ __forceinline__ v16h load_b16(const _Float16* __restrict__ col, int k0, int lh) {
  return *(const v16h*)(col + k0 + lh * 16);
}
__device__ __forceinline__ v8f wmma_f16(v16h a, v16h b, v8f c) {
  return __builtin_amdgcn_wmma_f32_16x16x32_f16(false, a, false, b, (short)0, c, false, false);
}

// ---------------------------------------------------------------------------
// CDNA5 async global->LDS copy (ASYNCcnt path), per-lane 16B.
// LDS operand = low 32 bits of the generic pointer (LDS aperture mapping).
// ---------------------------------------------------------------------------
__device__ __forceinline__ void async_ld_b128(const _Float16* g, _Float16* l) {
  asm volatile("global_load_async_to_lds_b128 %0, %1, off"
               :
               : "v"((unsigned)(size_t)l), "v"((unsigned long long)(size_t)g)
               : "memory");
}
__device__ __forceinline__ void wait_async0() {
  asm volatile("s_wait_asynccnt 0x0" ::: "memory");
}

// ---------------------------------------------------------------------------
// Precision conversion / transposes
// ---------------------------------------------------------------------------
__global__ void cvt_f32_f16(const float* __restrict__ in, _Float16* __restrict__ out, int n) {
  int i = blockIdx.x * 256 + threadIdx.x;
  if (i < n) out[i] = (_Float16)in[i];
}

// (H, M, D) f32 -> (H, D, M) f16
__global__ void transpose_w_qkv(const float* __restrict__ in, _Float16* __restrict__ out) {
  int i = blockIdx.x * 256 + threadIdx.x;  // H*M*D threads
  int d = i % D_;
  int m = (i / D_) % M_;
  int h = i / (D_ * M_);
  out[((size_t)h * D_ + d) * M_ + m] = (_Float16)in[i];
}

// (H*D, M) f32 -> (M, H*D) f16
__global__ void transpose_wao(const float* __restrict__ in, _Float16* __restrict__ out) {
  int i  = blockIdx.x * 256 + threadIdx.x;  // HD*M threads
  int m  = i % M_;
  int hd = i / M_;
  out[(size_t)m * HD_ + hd] = (_Float16)in[i];
}

// ---------------------------------------------------------------------------
// QKV projection: C[t, d] = sum_m X[t,m] * W[m,d]  (per head).
// Block = 4 waves x 16 tokens, all sharing one 128-wide weight panel.
// The 32(K) x 128(d) weight slab is async-staged into double-buffered LDS
// once per block (4x less global traffic); waves read B-fragments via DS.
// vtrans=0: out (B,H,T,D);  vtrans=1: out (B,H,D,T)  (V pre-transposed)
// ---------------------------------------------------------------------------
__global__ void __launch_bounds__(128) qkv_gemm(const _Float16* __restrict__ xh,
                                                const _Float16* __restrict__ wt,
                                                _Float16* __restrict__ out, int vtrans) {
  __shared__ __align__(32) _Float16 WT[2][128 * 32];  // 16 KB

  int tid  = threadIdx.x;
  int lane = tid & 31;
  int wave = tid >> 5;
  int lr = lane & 15, lh = lane >> 4;
  int t0 = (blockIdx.x * 4 + wave) * 16;
  int h  = blockIdx.y;

  const _Float16* wth  = wt + (size_t)h * D_ * M_;
  const _Float16* arow = xh + (size_t)(t0 + lr) * M_;

  // thread `tid` stages LDS row d = tid (32 f16 = 64 B = 4 async b128)
  const _Float16* grow = wth + (size_t)tid * M_;
  _Float16*       lrow = &WT[0][tid * 32];

  v8f z = {};
  v8f acc[8];
#pragma unroll
  for (int n = 0; n < 8; n++) acc[n] = z;

  // prologue: fill buffer 0 with K-slab m0 = 0
#pragma unroll
  for (int c = 0; c < 4; c++) async_ld_b128(grow + c * 8, lrow + c * 8);
  wait_async0();
  __syncthreads();

  for (int m0 = 0; m0 < M_; m0 += 32) {
    int nb = (m0 >> 5) & 1;
    if (m0 + 32 < M_) {
      const _Float16* g = grow + m0 + 32;
      _Float16*       l = &WT[nb ^ 1][tid * 32];
#pragma unroll
      for (int c = 0; c < 4; c++) async_ld_b128(g + c * 8, l + c * 8);
    }

    v16h a = load_a16(arow, m0, lh);
#pragma unroll
    for (int n = 0; n < 8; n++) {
      v16h b = *(const v16h*)(&WT[nb][(n * 16 + lr) * 32 + lh * 16]);
      acc[n] = wmma_f16(a, b, acc[n]);
    }

    wait_async0();   // next slab resident
    __syncthreads(); // all waves done reading current slab
  }

#pragma unroll
  for (int n = 0; n < 8; n++) {
#pragma unroll
    for (int r = 0; r < 8; r++) {
      int tg = t0 + r + lh * 8;        // global token (C-layout M index)
      int b  = tg >> 11;               // T_ == 2048
      int tl = tg & (T_ - 1);
      int d  = n * 16 + lr;            // C-layout N index
      size_t base = ((size_t)b * H_ + h) * (size_t)T_ * D_;
      size_t addr = vtrans ? base + (size_t)d * T_ + tl
                           : base + (size_t)tl * D_ + d;
      out[addr] = (_Float16)acc[n][r];
    }
  }
}

// ---------------------------------------------------------------------------
// RoPE + 1/sqrt(D) scaling, in-place on q and k (B,H,T,D) f16
// ---------------------------------------------------------------------------
__global__ void rope_scale(_Float16* __restrict__ q, _Float16* __restrict__ k) {
  int i  = blockIdx.x * 256 + threadIdx.x;  // B*H*T*HALF threads
  int d  = i & (HALF_ - 1);
  int t  = (i >> 6) & (T_ - 1);
  int bh = i >> 17;

  const float LOG_BASE = 9.210340371976184f;  // ln(10000)
  float ang = (float)t * __expf(-(float)d * (LOG_BASE / (float)HALF_));
  float s, c;
  __sincosf(ang, &s, &c);
  const float mult = 0.08838834764831845f;  // (1/D)^0.5

  size_t base = (size_t)bh * T_ * D_ + (size_t)t * D_;
  float qe = (float)q[base + d], qo = (float)q[base + d + HALF_];
  q[base + d]         = (_Float16)((qe * c - qo * s) * mult);
  q[base + d + HALF_] = (_Float16)((qe * s + qo * c) * mult);
  float ke = (float)k[base + d], ko = (float)k[base + d + HALF_];
  k[base + d]         = (_Float16)((ke * c - ko * s) * mult);
  k[base + d + HALF_] = (_Float16)((ke * s + ko * c) * mult);
}

// ---------------------------------------------------------------------------
// Flash attention (causal). One wave owns 16 query rows of one (b,h).
// Computes S^T = K_tile x Q^T so softmax row stats are per-lane (i = lane&15).
// P^T round-trips through LDS (row-major by i) to land in WMMA B-layout.
// O accumulated transposed: o^T[d,i];  V read from (B,H,D,T) layout.
// ---------------------------------------------------------------------------
__global__ void __launch_bounds__(128) attn_fwd(const _Float16* __restrict__ q,
                                                const _Float16* __restrict__ k,
                                                const _Float16* __restrict__ vt,
                                                _Float16* __restrict__ o) {
  __shared__ __align__(32) _Float16 P2[4][16][32];

  int lane = threadIdx.x & 31;
  int wave = threadIdx.x >> 5;
  int lr = lane & 15, lh = lane >> 4;
  int h = blockIdx.y, b = blockIdx.z;
  int i0 = (blockIdx.x * 4 + wave) * 16;

  size_t hb = ((size_t)b * H_ + h) * (size_t)T_ * D_;
  const _Float16* qh = q + hb;
  const _Float16* kh = k + hb;
  const _Float16* vh = vt + hb;

  // Q^T as loop-invariant B-fragments: lane holds column i = i0+lr
  v16h qb[4];
#pragma unroll
  for (int kk = 0; kk < 4; kk++)
    qb[kk] = load_b16(qh + (size_t)(i0 + lr) * D_, kk * 32, lh);

  v8f z = {};
  v8f acc[8];
#pragma unroll
  for (int n = 0; n < 8; n++) acc[n] = z;

  float m_i = -3.0e38f, l_i = 0.0f;
  int ig = i0 + lr;

  for (int j0 = 0; j0 <= i0; j0 += 32) {
    // --- S^T tiles: st[t][r] = s[j0 + t*16 + r + lh*8, i0 + lr]
    v8f st[2];
#pragma unroll
    for (int t = 0; t < 2; t++) {
      int jb = j0 + t * 16;
      v8f s = z;
#pragma unroll
      for (int kk = 0; kk < 4; kk++) {
        v16h ka = load_a16(kh + (size_t)(jb + lr) * D_, kk * 32, lh);
        s = wmma_f16(ka, qb[kk], s);
      }
      st[t] = s;
    }

    // --- causal mask + row max
    float mt = -3.0e38f;
#pragma unroll
    for (int t = 0; t < 2; t++) {
#pragma unroll
      for (int r = 0; r < 8; r++) {
        int jg = j0 + t * 16 + lh * 8 + r;
        float sv = st[t][r];
        sv = (jg > ig) ? -1.0e30f : sv;
        st[t][r] = sv;
        mt = fmaxf(mt, sv);
      }
    }
    mt = fmaxf(mt, __shfl_xor(mt, 16, 32));
    float m_new = fmaxf(m_i, mt);
    float ef = __expf(m_i - m_new);
    m_i = m_new;

    // --- P = exp(S - m), stage P^T into LDS in B-layout order
    float ps = 0.0f;
#pragma unroll
    for (int t = 0; t < 2; t++) {
      v8h pf;
#pragma unroll
      for (int r = 0; r < 8; r++) {
        float p = __expf(st[t][r] - m_new);
        ps += p;
        pf[r] = (_Float16)p;
      }
      *(v8h*)(&P2[wave][lr][t * 16 + lh * 8]) = pf;  // row i=lr, cols j-local
    }
    ps += __shfl_xor(ps, 16, 32);
    l_i = l_i * ef + ps;

#pragma unroll
    for (int n = 0; n < 8; n++)
#pragma unroll
      for (int r = 0; r < 8; r++) acc[n][r] *= ef;

    // --- O^T += V^T x P^T  (same-wave LDS store->load is ordered; no barrier)
    v16h pb = *(const v16h*)(&P2[wave][lr][lh * 16]);
#pragma unroll
    for (int n = 0; n < 8; n++) {
      v16h va = load_a16(vh + (size_t)(n * 16 + lr) * T_, j0, lh);
      acc[n] = wmma_f16(va, pb, acc[n]);
    }
  }

  // --- normalize and store o (B,T,H,D): i = i0+lr, d = n*16 + r + lh*8
  float inv = 1.0f / l_i;
#pragma unroll
  for (int n = 0; n < 8; n++) {
#pragma unroll
    for (int r = 0; r < 8; r++) {
      int dloc = n * 16 + r + lh * 8;
      o[(((size_t)b * T_ + (i0 + lr)) * H_ + h) * (size_t)D_ + dloc] =
          (_Float16)(acc[n][r] * inv);
    }
  }
}

// ---------------------------------------------------------------------------
// Output projection: R[t, m] = sum_hd O[t, hd] * Wao[hd, m],  f32 out.
// Same async-LDS double-buffered staging of the 32(K) x 128(m) weight slab.
// ---------------------------------------------------------------------------
__global__ void __launch_bounds__(128) out_proj(const _Float16* __restrict__ oh,
                                                const _Float16* __restrict__ waoT,
                                                float* __restrict__ out) {
  __shared__ __align__(32) _Float16 WOT[2][128 * 32];  // 16 KB

  int tid  = threadIdx.x;
  int lane = tid & 31;
  int wave = tid >> 5;
  int lr = lane & 15, lh = lane >> 4;
  int t0 = (blockIdx.x * 4 + wave) * 16;
  int m0 = blockIdx.y * 128;

  const _Float16* arow = oh + (size_t)(t0 + lr) * HD_;
  const _Float16* grow = waoT + (size_t)(m0 + tid) * HD_;  // thread stages m-row tid

  v8f z = {};
  v8f acc[8];
#pragma unroll
  for (int n = 0; n < 8; n++) acc[n] = z;

#pragma unroll
  for (int c = 0; c < 4; c++) async_ld_b128(grow + c * 8, &WOT[0][tid * 32] + c * 8);
  wait_async0();
  __syncthreads();

  for (int k0 = 0; k0 < HD_; k0 += 32) {
    int nb = (k0 >> 5) & 1;
    if (k0 + 32 < HD_) {
      const _Float16* g = grow + k0 + 32;
      _Float16*       l = &WOT[nb ^ 1][tid * 32];
#pragma unroll
      for (int c = 0; c < 4; c++) async_ld_b128(g + c * 8, l + c * 8);
    }

    v16h a = load_a16(arow, k0, lh);
#pragma unroll
    for (int n = 0; n < 8; n++) {
      v16h bf = *(const v16h*)(&WOT[nb][(n * 16 + lr) * 32 + lh * 16]);
      acc[n] = wmma_f16(a, bf, acc[n]);
    }

    wait_async0();
    __syncthreads();
  }

#pragma unroll
  for (int n = 0; n < 8; n++) {
#pragma unroll
    for (int r = 0; r < 8; r++) {
      int tg = t0 + r + lh * 8;
      int mc = m0 + n * 16 + lr;
      out[(size_t)tg * M_ + mc] = acc[n][r];
    }
  }
}

// ---------------------------------------------------------------------------
extern "C" void kernel_launch(void* const* d_in, const int* in_sizes, int n_in,
                              void* d_out, int out_size, void* d_ws, size_t ws_size,
                              hipStream_t stream) {
  (void)in_sizes; (void)n_in; (void)out_size; (void)ws_size;
  const float* x    = (const float*)d_in[0];
  const float* w_aq = (const float*)d_in[1];
  const float* w_ak = (const float*)d_in[2];
  const float* w_av = (const float*)d_in[3];
  const float* w_ao = (const float*)d_in[4];
  float* out = (float*)d_out;

  char* ws = (char*)d_ws;
  size_t off = 0;
  auto alloc = [&](size_t bytes) -> void* {
    void* p = ws + off;
    off += (bytes + 255) & ~(size_t)255;
    return p;
  };
  _Float16* xh   = (_Float16*)alloc((size_t)BT_ * M_ * 2);
  _Float16* wqh  = (_Float16*)alloc((size_t)H_ * D_ * M_ * 2);
  _Float16* wkh  = (_Float16*)alloc((size_t)H_ * D_ * M_ * 2);
  _Float16* wvh  = (_Float16*)alloc((size_t)H_ * D_ * M_ * 2);
  _Float16* waoT = (_Float16*)alloc((size_t)M_ * HD_ * 2);
  _Float16* qd   = (_Float16*)alloc((size_t)B_ * H_ * T_ * D_ * 2);
  _Float16* kd   = (_Float16*)alloc((size_t)B_ * H_ * T_ * D_ * 2);
  _Float16* vtd  = (_Float16*)alloc((size_t)B_ * H_ * T_ * D_ * 2);
  _Float16* od   = (_Float16*)alloc((size_t)BT_ * HD_ * 2);

  cvt_f32_f16<<<(BT_ * M_) / 256, 256, 0, stream>>>(x, xh, BT_ * M_);
  transpose_w_qkv<<<(H_ * M_ * D_) / 256, 256, 0, stream>>>(w_aq, wqh);
  transpose_w_qkv<<<(H_ * M_ * D_) / 256, 256, 0, stream>>>(w_ak, wkh);
  transpose_w_qkv<<<(H_ * M_ * D_) / 256, 256, 0, stream>>>(w_av, wvh);
  transpose_wao<<<(HD_ * M_) / 256, 256, 0, stream>>>(w_ao, waoT);

  dim3 gq(BT_ / 64, H_);
  qkv_gemm<<<gq, 128, 0, stream>>>(xh, wqh, qd, 0);
  qkv_gemm<<<gq, 128, 0, stream>>>(xh, wkh, kd, 0);
  qkv_gemm<<<gq, 128, 0, stream>>>(xh, wvh, vtd, 1);

  rope_scale<<<(B_ * H_ * T_ * HALF_) / 256, 256, 0, stream>>>(qd, kd);

  attn_fwd<<<dim3(T_ / 64, H_, B_), 128, 0, stream>>>(qd, kd, vtd, od);

  out_proj<<<dim3(BT_ / 64, M_ / 128), 128, 0, stream>>>(od, waoT, out);
}
